// GatedGCN_19464791785727
// MI455X (gfx1250) — compile-verified, matrix operand order checked by reading, not run on hardware
//
#include <hip/hip_runtime.h>

// ---------------------------------------------------------------------------
// GatedGCN forward for MI455X (gfx1250, wave32, WMMA 16x16x32 f16)
//   h = x @ Wenc + b_enc
//   4x: fused QKVS gemm (A-fragments loaded once, 4 weights from LDS):
//         k/q/v (f16) ; aggr = h@Ws + b (f32, skip pre-seeded)
//       edge: eta = sigmoid(k[dst]+q[src]); atomicAdd(aggr[dst], eta*v[src])
//       h = leaky(aggr) + h ; h16 = (f16)h
//   mean-pool over batch -> d_out[128*128]
// ---------------------------------------------------------------------------

#define DF 128
#define NNODES 100000
#define NEDGES 600000
#define NGRAPH 128

typedef _Float16 half_t;
typedef __attribute__((ext_vector_type(16))) _Float16 v16h;
typedef __attribute__((ext_vector_type(8)))  _Float16 v8h;
typedef __attribute__((ext_vector_type(4)))  _Float16 v4h;
typedef __attribute__((ext_vector_type(8)))  float    v8f;

// 16x32 f16 WMMA fragment for this lane from a row-major slab:
// lanes 0-15 hold K = [0..7],[16..23]; lanes 16-31 hold K = [8..15],[24..31].
__device__ __forceinline__ v16h load_frag(const half_t* p) {
    union { v16h v; v8h h[2]; } u;
    u.h[0] = *(const v8h*)(p);
    u.h[1] = *(const v8h*)(p + 16);
    return u.v;
}

// ---------------------------------------------------------------------------
// Weight prep: f32 [128x128] -> f16 TRANSPOSED (BT[n][k] = W[k][n]).
// Layout: slot 0 = enc; slot 1+4L+{0,1,2,3} = {Wk,Wq,Wv,Ws}[L]  (fused-friendly)
// ---------------------------------------------------------------------------
__global__ __launch_bounds__(256)
void prep_weights(const float* __restrict__ Wenc, const float* __restrict__ Wk,
                  const float* __restrict__ Wq,   const float* __restrict__ Wv,
                  const float* __restrict__ Ws,   half_t* __restrict__ out) {
    int idx = blockIdx.x * 256 + threadIdx.x;
    if (idx >= 17 * DF * DF) return;
    int slot = idx / (DF * DF);
    int r    = idx % (DF * DF);
    int n = r / DF, k = r % DF;
    const float* src;
    if (slot == 0) {
        src = Wenc;
    } else {
        int t = slot - 1, L = t >> 2, which = t & 3;
        const float* base = (which == 0) ? Wk : (which == 1) ? Wq : (which == 2) ? Wv : Ws;
        src = base + (size_t)L * DF * DF;
    }
    out[idx] = (half_t)src[(size_t)k * DF + n];
}

__global__ __launch_bounds__(256)
void f32_to_f16_v4(const float4* __restrict__ in, v4h* __restrict__ out, int n4) {
    int i = blockIdx.x * 256 + threadIdx.x;
    if (i >= n4) return;
    float4 v = in[i];
    v4h o; o[0] = (half_t)v.x; o[1] = (half_t)v.y; o[2] = (half_t)v.z; o[3] = (half_t)v.w;
    out[i] = o;
}

// ---------------------------------------------------------------------------
// Encoder GEMM: h(f32) and h16(f16) = x16 @ Wenc + b_enc. One 32KB LDS panel.
// ---------------------------------------------------------------------------
template<bool BOUNDS>
__global__ __launch_bounds__(256)
void gemm_enc(const half_t* __restrict__ A, const half_t* __restrict__ BT,
              const float* __restrict__ bias, float* __restrict__ outF,
              half_t* __restrict__ outH, int rowOff, int M) {
    __shared__ __align__(16) half_t sB[DF * DF];
    const int tid = threadIdx.x;
    {
        const v8h* s = (const v8h*)BT;
        v8h*       d = (v8h*)sB;
        #pragma unroll
        for (int i = 0; i < (DF * DF / 8) / 256; ++i) d[tid + i * 256] = s[tid + i * 256];
    }
    __syncthreads();

    const int wave  = tid >> 5;
    const int lane  = tid & 31;
    const int rlane = lane & 15;
    const int koff  = (lane & 16) ? 8 : 0;
    const int rowTile = rowOff + blockIdx.x * 128 + wave * 16;

    int rowg = rowTile + rlane;
    if (BOUNDS && rowg >= M) rowg = M - 1;

    v16h afrag[4];
    #pragma unroll
    for (int kt = 0; kt < 4; ++kt)
        afrag[kt] = load_frag(A + (size_t)rowg * DF + kt * 32 + koff);

    const int rbase = rowTile + ((lane & 16) ? 8 : 0);

    #pragma unroll
    for (int nt = 0; nt < 8; ++nt) {
        v8f acc = {};
        #pragma unroll
        for (int kt = 0; kt < 4; ++kt) {
            v16h bfrag = load_frag(sB + (size_t)(nt * 16 + rlane) * DF + kt * 32 + koff);
            acc = __builtin_amdgcn_wmma_f32_16x16x32_f16(
                      false, afrag[kt], false, bfrag, (short)0, acc, false, false);
        }
        const int col = nt * 16 + rlane;
        const float bv = bias[col];
        #pragma unroll
        for (int r = 0; r < 8; ++r) {
            int row = rbase + r;
            if (!BOUNDS || row < M) {
                float v = acc[r] + bv;
                outF[(size_t)row * DF + col] = v;
                outH[(size_t)row * DF + col] = (half_t)v;
            }
        }
    }
}

// ---------------------------------------------------------------------------
// Fused per-layer GEMM: A-fragments loaded ONCE, 4 weights (k,q,v,s) applied.
//   wT4 = transposed f16 weights, contiguous [4][128][128] = {Wk,Wq,Wv,Ws}[L].
//   k/q/v written f16 (L2-resident gather set); skip+bias seeded into aggr f32.
//   LDS double-staged 2x32KB; 128 v_wmma per wave.
// ---------------------------------------------------------------------------
template<bool BOUNDS>
__global__ __launch_bounds__(256)
void gemm_qkvs(const half_t* __restrict__ A, const half_t* __restrict__ wT4,
               const float* __restrict__ bk, const float* __restrict__ bq,
               const float* __restrict__ bv, const float* __restrict__ bcv,
               half_t* __restrict__ kb, half_t* __restrict__ qb,
               half_t* __restrict__ vb, float* __restrict__ aggr,
               int rowOff, int M) {
    __shared__ __align__(16) half_t sB[2 * DF * DF];   // 64 KB, two weight panels
    const int tid   = threadIdx.x;
    const int wave  = tid >> 5;
    const int lane  = tid & 31;
    const int rlane = lane & 15;
    const int koff  = (lane & 16) ? 8 : 0;
    const int rowTile = rowOff + blockIdx.x * 128 + wave * 16;

    int rowg = rowTile + rlane;
    if (BOUNDS && rowg >= M) rowg = M - 1;

    v16h afrag[4];
    #pragma unroll
    for (int kt = 0; kt < 4; ++kt)
        afrag[kt] = load_frag(A + (size_t)rowg * DF + kt * 32 + koff);

    const int rbase = rowTile + ((lane & 16) ? 8 : 0);
    half_t* const outs[3]   = { kb, qb, vb };
    const float* const bias[4] = { bk, bq, bv, bcv };

    #pragma unroll
    for (int stage = 0; stage < 2; ++stage) {
        if (stage != 0) __syncthreads();               // panel reuse: drain readers
        {
            const v8h* s = (const v8h*)(wT4 + (size_t)stage * 2 * DF * DF);
            v8h*       d = (v8h*)sB;
            #pragma unroll
            for (int i = 0; i < (2 * DF * DF / 8) / 256; ++i)
                d[tid + i * 256] = s[tid + i * 256];
        }
        __syncthreads();

        #pragma unroll
        for (int wi = 0; wi < 2; ++wi) {
            const int w = stage * 2 + wi;
            const half_t* sBw = sB + (size_t)wi * DF * DF;
            #pragma unroll
            for (int nt = 0; nt < 8; ++nt) {
                v8f acc = {};
                #pragma unroll
                for (int kt = 0; kt < 4; ++kt) {
                    v16h bfrag = load_frag(sBw + (size_t)(nt * 16 + rlane) * DF +
                                           kt * 32 + koff);
                    acc = __builtin_amdgcn_wmma_f32_16x16x32_f16(
                              false, afrag[kt], false, bfrag, (short)0, acc, false, false);
                }
                const int col = nt * 16 + rlane;
                const float bvs = bias[w][col];
                #pragma unroll
                for (int r = 0; r < 8; ++r) {
                    int row = rbase + r;
                    if (!BOUNDS || row < M) {
                        float v = acc[r] + bvs;
                        if (w == 3) aggr[(size_t)row * DF + col] = v;
                        else        outs[w][(size_t)row * DF + col] = (half_t)v;
                    }
                }
            }
        }
    }
}

// ---------------------------------------------------------------------------
// Edge gather/gate/scatter: 32 lanes per edge, 4 feats per lane (f16 in, L2-hot)
// ---------------------------------------------------------------------------
__global__ __launch_bounds__(256)
void edge_gate(const long long* __restrict__ srcIdx, const long long* __restrict__ dstIdx,
               const half_t* __restrict__ kb, const half_t* __restrict__ qb,
               const half_t* __restrict__ vb, float* __restrict__ aggr, int E) {
    int e    = blockIdx.x * 8 + (threadIdx.x >> 5);
    int lane = threadIdx.x & 31;
    if (e >= E) return;
    long long i = dstIdx[e];   // target (aggregation index)
    long long j = srcIdx[e];   // source
    v4h kv = ((const v4h*)(kb + (size_t)i * DF))[lane];
    v4h qv = ((const v4h*)(qb + (size_t)j * DF))[lane];
    v4h vv = ((const v4h*)(vb + (size_t)j * DF))[lane];
    float* ap = aggr + (size_t)i * DF + lane * 4;
    #pragma unroll
    for (int c = 0; c < 4; ++c) {
        float g   = (float)kv[c] + (float)qv[c];
        float eta = 1.0f / (1.0f + __expf(-g));
        atomicAdd(ap + c, eta * (float)vv[c]);
    }
}

// h = leaky(aggr) + h ; h16 = (f16)h     (aggr already = msgsum + h@Ws + b)
__global__ __launch_bounds__(256)
void layer_epilogue(const float4* __restrict__ aggr, float4* __restrict__ h,
                    v4h* __restrict__ h16, int n4) {
    int i = blockIdx.x * 256 + threadIdx.x;
    if (i >= n4) return;
    float4 a = aggr[i];
    a.x = (a.x >= 0.0f) ? a.x : 0.01f * a.x;
    a.y = (a.y >= 0.0f) ? a.y : 0.01f * a.y;
    a.z = (a.z >= 0.0f) ? a.z : 0.01f * a.z;
    a.w = (a.w >= 0.0f) ? a.w : 0.01f * a.w;
    float4 hv = h[i];
    hv.x += a.x; hv.y += a.y; hv.z += a.z; hv.w += a.w;
    h[i] = hv;
    v4h o; o[0] = (half_t)hv.x; o[1] = (half_t)hv.y; o[2] = (half_t)hv.z; o[3] = (half_t)hv.w;
    h16[i] = o;
}

// ---------------------------------------------------------------------------
// Global mean pool
// ---------------------------------------------------------------------------
__global__ __launch_bounds__(256)
void pool_zero(float* __restrict__ sums, float* __restrict__ counts) {
    int i = blockIdx.x * 256 + threadIdx.x;
    if (i < NGRAPH * DF) sums[i] = 0.0f;
    if (i < NGRAPH)      counts[i] = 0.0f;
}

__global__ __launch_bounds__(256)
void pool_accum(const float* __restrict__ h, const long long* __restrict__ batch,
                float* __restrict__ sums, float* __restrict__ counts, int M) {
    int n    = blockIdx.x * 8 + (threadIdx.x >> 5);
    int lane = threadIdx.x & 31;
    if (n >= M) return;
    long long g = batch[n];
    float4 v = ((const float4*)(h + (size_t)n * DF))[lane];
    float* sp = sums + (size_t)g * DF + lane * 4;
    atomicAdd(sp + 0, v.x); atomicAdd(sp + 1, v.y);
    atomicAdd(sp + 2, v.z); atomicAdd(sp + 3, v.w);
    if (lane == 0) atomicAdd(counts + g, 1.0f);
}

__global__ __launch_bounds__(256)
void pool_final(const float* __restrict__ sums, const float* __restrict__ counts,
                float* __restrict__ out, int out_size) {
    int i = blockIdx.x * 256 + threadIdx.x;
    if (i < NGRAPH * DF) {
        float c = counts[i / DF];
        out[i] = sums[i] / (c > 1.0f ? c : 1.0f);
    }
    if (i == 0 && out_size > NGRAPH * DF) out[NGRAPH * DF] = 0.0f;  // tuple's scalar 0
}

// ---------------------------------------------------------------------------
extern "C" void kernel_launch(void* const* d_in, const int* in_sizes, int n_in,
                              void* d_out, int out_size, void* d_ws, size_t ws_size,
                              hipStream_t stream) {
    (void)in_sizes; (void)n_in;
    const float*     x     = (const float*)d_in[0];
    const long long* ei    = (const long long*)d_in[1];
    const long long* batch = (const long long*)d_in[2];
    const float* Wenc = (const float*)d_in[3];
    const float* benc = (const float*)d_in[4];
    const float* Wk   = (const float*)d_in[5];
    const float* bk   = (const float*)d_in[6];
    const float* Wq   = (const float*)d_in[7];
    const float* bq   = (const float*)d_in[8];
    const float* Wv   = (const float*)d_in[9];
    const float* bv   = (const float*)d_in[10];
    const float* Ws   = (const float*)d_in[11];
    const float* bcv  = (const float*)d_in[12];

    const size_t NM = (size_t)NNODES * DF;  // 12.8M elems
    char* ws = (char*)d_ws;
    size_t off = 0;
    float*  h    = (float*) (ws + off); off += NM * 4;
    half_t* h16  = (half_t*)(ws + off); off += NM * 2;
    half_t* x16  = (half_t*)(ws + off); off += NM * 2;
    half_t* kb   = (half_t*)(ws + off); off += NM * 2;
    half_t* qb   = (half_t*)(ws + off); off += NM * 2;
    half_t* vb   = (half_t*)(ws + off); off += NM * 2;
    float*  aggr = (float*) (ws + off); off += NM * 4;
    half_t* wT   = (half_t*)(ws + off); off += (size_t)17 * DF * DF * 2;
    float*  psum = (float*) (ws + off); off += (size_t)NGRAPH * DF * 4;
    float*  pcnt = (float*) (ws + off); off += (size_t)NGRAPH * 4;
    if (ws_size < off) return;

    const int fullBlocks = NNODES / 128;               // 781 (no bounds checks)
    const int tailOff    = fullBlocks * 128;           // 99968 (one guarded block)
    const int elemBlocks4 = (int)((NM / 4 + 255) / 256);

    prep_weights<<<(17 * DF * DF + 255) / 256, 256, 0, stream>>>(Wenc, Wk, Wq, Wv, Ws, wT);
    f32_to_f16_v4<<<elemBlocks4, 256, 0, stream>>>((const float4*)x, (v4h*)x16,
                                                   (int)(NM / 4));

    gemm_enc<false><<<fullBlocks, 256, 0, stream>>>(x16, wT, benc, h, h16, 0, NNODES);
    gemm_enc<true><<<1, 256, 0, stream>>>(x16, wT, benc, h, h16, tailOff, NNODES);

    for (int L = 0; L < 4; ++L) {
        const half_t* wT4 = wT + (size_t)(1 + 4 * L) * DF * DF;   // {Wk,Wq,Wv,Ws}[L]
        gemm_qkvs<false><<<fullBlocks, 256, 0, stream>>>(
            h16, wT4, bk + L * DF, bq + L * DF, bv + L * DF, bcv + L * DF,
            kb, qb, vb, aggr, 0, NNODES);
        gemm_qkvs<true><<<1, 256, 0, stream>>>(
            h16, wT4, bk + L * DF, bq + L * DF, bv + L * DF, bcv + L * DF,
            kb, qb, vb, aggr, tailOff, NNODES);
        edge_gate<<<(NEDGES + 7) / 8, 256, 0, stream>>>(ei, ei + NEDGES, kb, qb, vb,
                                                        aggr, NEDGES);
        layer_epilogue<<<elemBlocks4, 256, 0, stream>>>((const float4*)aggr, (float4*)h,
                                                        (v4h*)h16, (int)(NM / 4));
    }

    pool_zero<<<(NGRAPH * DF + 255) / 256, 256, 0, stream>>>(psum, pcnt);
    pool_accum<<<(NNODES + 7) / 8, 256, 0, stream>>>(h, batch, psum, pcnt, NNODES);
    pool_final<<<(NGRAPH * DF + 255) / 256, 256, 0, stream>>>(psum, pcnt, (float*)d_out,
                                                              out_size);
}
